// GraphZ_57320633532487
// MI455X (gfx1250) — matrix operand order repaired and checked by reading, not compiled, not run on hardware
//
#include <hip/hip_runtime.h>
#include <stdint.h>

// ---------------------------------------------------------------------------
// GraphConv GNN (4 layers) for MI455X / gfx1250, wave32 + WMMA f16->f32.
//   per layer: h_rel = Xh @ Wrel^T ; acc = Xh @ Wroot^T + b   (dual WMMA GEMM)
//              acc[dst] += h_rel[src]  over edges             (f32 atomics, L2)
//              X_next = relu(acc) cast to f16 (padded)        (elementwise)
//   final: log_softmax over 2 classes.
// GEMM A-panels are DMA'd global->LDS with GLOBAL_LOAD_ASYNC_TO_LDS_B64
// (ASYNCcnt), triple-buffered, one barrier per 64-wide K step, 4 WMMAs/step.
// ---------------------------------------------------------------------------

typedef __attribute__((ext_vector_type(16))) _Float16 v16h;
typedef __attribute__((ext_vector_type(8)))  float    v8f;

union ABFrag { v16h v; uint4 q[2]; };

#define N_NODES 50000
#define N_EDGES 800000

// CDNA5 async global->LDS DMA (tracked by ASYNCcnt). vdst = LDS byte address
// (low 32 bits of the generic shared pointer), vaddr = 64-bit global address.
__device__ __forceinline__ void async_load_b64_to_lds(uint32_t lds_off, uint64_t gaddr)
{
    asm volatile("global_load_async_to_lds_b64 %0, %1, off"
                 :: "v"(lds_off), "v"(gaddr)
                 : "memory");
}

// ---------------------------------------------------------------------------
// Dual GEMM: one block = one 16-row M tile; 8 waves = up to 8 N tiles.
// Per iteration: 16x64 f16 A panel arrives in LDS via async DMA issued one
// iteration ahead; each wave builds two A fragments and issues
// 4 x v_wmma_f32_16x16x32_f16 (rel+root, two K chunks).
// Buffer rotation is an explicit wrap-around counter (no % -> no magic-number
// division in the inner loop). Wave-activity guard is forced scalar via
// readfirstlane so EXEC stays all-1s around WMMA (ISA requirement).
// ---------------------------------------------------------------------------
__global__ __launch_bounds__(256)
void wmma_dual_gemm(const _Float16* __restrict__ X,      // [M x Kp] f16, padded
                    const _Float16* __restrict__ Brel,   // [Np x Kp] f16, padded
                    const _Float16* __restrict__ Broot,  // [Np x Kp] f16, padded
                    const float*    __restrict__ biasP,  // [Np] f32, padded
                    float*          __restrict__ Hrel,   // [M x Np] f32
                    float*          __restrict__ Acc,    // [M x Np] f32
                    int Kp, int Np, int nTiles, int kIters)   // kIters = Kp/64
{
    // 3 panels x 16 rows x 64 halves = 3 x 2KB, buffer stride 2048B (buf<<11)
    __shared__ __align__(16) _Float16 As[3][16 * 64];

    const int tid  = threadIdx.x;
    const int wave = tid >> 5;          // 0..7
    const int lane = tid & 31;          // wave32
    const int rowBase = blockIdx.x * 16;

    // wave-uniform tile id, forced into an SGPR -> scalar branch, EXEC untouched
    const int  nTile  = __builtin_amdgcn_readfirstlane(blockIdx.y * 8 + wave);
    const bool active = (nTile < nTiles);
    const int  nBase  = nTile * 16;

    const int r = lane & 15;            // A row / B column / C column
    const int h = lane >> 4;            // lane-half selector

    // A staging: 256 threads DMA 16 rows x 64 halves (one b64 each)
    const int sr = tid >> 4;            // 0..15 row
    const int sd = (tid & 15) * 4;      // half offset 0,4,..,60
    const _Float16* xrow = X + (long long)(rowBase + sr) * Kp + sd;
    const uint32_t  ldsWrBase = (uint32_t)(uintptr_t)&As[0][sr * 64 + sd];
    const _Float16* ldsRdBase = &As[0][r * 64 + h * 8];   // + buf*1024 halves

    // B per-lane base: column (nBase+r), halves 0..7 = K[8h..], 8..15 = K[16+8h..]
    const _Float16* brelRow  = active ? (Brel  + (long long)(nBase + r) * Kp + h * 8) : Brel;
    const _Float16* brootRow = active ? (Broot + (long long)(nBase + r) * Kp + h * 8) : Broot;

    v8f crel  = {};
    v8f croot = {};

    // prologue: DMA panel 0 into buffer 0
    async_load_b64_to_lds(ldsWrBase, (uint64_t)(uintptr_t)xrow);

    int bufCur = 0;
    for (int kc = 0; kc < kIters; ++kc) {
        const int k0 = kc * 64;
        int bufNxt = bufCur + 1;
        if (bufNxt == 3) bufNxt = 0;                     // uniform s_cselect

        if (kc + 1 < kIters) {
            // DMA next panel while this one is consumed
            async_load_b64_to_lds(ldsWrBase + ((uint32_t)bufNxt << 11),
                                  (uint64_t)(uintptr_t)(xrow + k0 + 64));
            // next weight chunk into cache (global_prefetch_b8)
            __builtin_prefetch(brelRow  + k0 + 64, 0, 3);
            __builtin_prefetch(brootRow + k0 + 64, 0, 3);
            asm volatile("s_wait_asynccnt 1" ::: "memory");   // panel kc landed
        } else {
            asm volatile("s_wait_asynccnt 0" ::: "memory");
        }
        __syncthreads();

        // A fragments (16-bit A 16x32): halves 0..7 = K[8h..], 8..15 = K[16+8h..]
        const _Float16* panel = ldsRdBase + bufCur * 1024;
        ABFrag a0, a1;
        a0.q[0] = *(const uint4*)(panel);
        a0.q[1] = *(const uint4*)(panel + 16);
        a1.q[0] = *(const uint4*)(panel + 32);
        a1.q[1] = *(const uint4*)(panel + 48);

        if (active) {
            ABFrag brel0, brel1, broot0, broot1;
            brel0.q[0]  = *(const uint4*)(brelRow  + k0);
            brel0.q[1]  = *(const uint4*)(brelRow  + k0 + 16);
            brel1.q[0]  = *(const uint4*)(brelRow  + k0 + 32);
            brel1.q[1]  = *(const uint4*)(brelRow  + k0 + 48);
            broot0.q[0] = *(const uint4*)(brootRow + k0);
            broot0.q[1] = *(const uint4*)(brootRow + k0 + 16);
            broot1.q[0] = *(const uint4*)(brootRow + k0 + 32);
            broot1.q[1] = *(const uint4*)(brootRow + k0 + 48);
            crel  = __builtin_amdgcn_wmma_f32_16x16x32_f16(false, a0.v, false, brel0.v,
                                                           (short)0, crel,  false, false);
            croot = __builtin_amdgcn_wmma_f32_16x16x32_f16(false, a0.v, false, broot0.v,
                                                           (short)0, croot, false, false);
            crel  = __builtin_amdgcn_wmma_f32_16x16x32_f16(false, a1.v, false, brel1.v,
                                                           (short)0, crel,  false, false);
            croot = __builtin_amdgcn_wmma_f32_16x16x32_f16(false, a1.v, false, broot1.v,
                                                           (short)0, croot, false, false);
        }
        bufCur = bufNxt;
        // no trailing barrier: triple buffering + the pre-compute barrier keep
        // DMA writes (iter i -> next buffer, last read at iter i-2) race-free
    }

    if (active) {
        const int   col  = nBase + r;
        const float bv   = biasP[col];
        const int   row0 = rowBase + h * 8;     // C/D: lanes 16..31 hold M=8..15
        #pragma unroll
        for (int v = 0; v < 8; ++v) {
            const long long o = (long long)(row0 + v) * Np + col;
            Hrel[o] = crel[v];
            Acc[o]  = croot[v] + bv;
        }
    }
}

// ---------------------------------------------------------------------------
// Edge scatter-add, wide rows: one block per edge, contiguous row accesses,
// f32 atomics (global_atomic_add_f32 — L2-resident working set).
// ---------------------------------------------------------------------------
__global__ __launch_bounds__(256)
void scatter_add_kernel(const float* __restrict__ hrel, float* __restrict__ acc,
                        const long long* __restrict__ srcIdx,
                        const long long* __restrict__ dstIdx,
                        int width, int stride)
{
    const int e = blockIdx.x;
    const int s = (int)srcIdx[e];
    const int d = (int)dstIdx[e];
    const float* from = hrel + (long long)s * stride;
    float*       to   = acc  + (long long)d * stride;
    for (int j = threadIdx.x; j < width; j += blockDim.x)
        atomicAdd(&to[j], from[j]);
}

// width == 2 specialization: thread per (edge, component), shifts only
__global__ __launch_bounds__(256)
void scatter_add_w2_kernel(const float* __restrict__ hrel, float* __restrict__ acc,
                           const long long* __restrict__ srcIdx,
                           const long long* __restrict__ dstIdx,
                           int stride, int nEdges)
{
    const int t = blockIdx.x * blockDim.x + threadIdx.x;
    const int e = t >> 1;
    if (e >= nEdges) return;
    const int j = t & 1;
    const int s = (int)srcIdx[e];
    const int d = (int)dstIdx[e];
    atomicAdd(&acc[(long long)d * stride + j], hrel[(long long)s * stride + j]);
}

// ---------------------------------------------------------------------------
// Pad + cast f32 -> f16 with optional ReLU. 2D grid: blockIdx.y = row,
// x-dim covers dstStride columns. No integer division.
// ---------------------------------------------------------------------------
__global__ __launch_bounds__(256)
void cast_pad_kernel(const float* __restrict__ src, int srcStride, int width, int rowLimit,
                     _Float16* __restrict__ dst, int dstStride, int doRelu)
{
    const int n = blockIdx.y;
    const int k = blockIdx.x * blockDim.x + threadIdx.x;
    if (k >= dstStride) return;
    float v = 0.0f;
    if (n < rowLimit && k < width) {
        v = src[(long long)n * srcStride + k];
        if (doRelu && v < 0.0f) v = 0.0f;
    }
    dst[(long long)n * dstStride + k] = (_Float16)v;
}

__global__ void bias_pad_kernel(const float* __restrict__ src, float* __restrict__ dst,
                                int N, int Np)
{
    const int i = blockIdx.x * blockDim.x + threadIdx.x;
    if (i < Np) dst[i] = (i < N) ? src[i] : 0.0f;
}

__global__ __launch_bounds__(256)
void logsoftmax2_kernel(const float* __restrict__ acc, float* __restrict__ out,
                        int stride, int nRows)
{
    const int i = blockIdx.x * blockDim.x + threadIdx.x;
    if (i >= nRows) return;
    const float a0 = acc[(long long)i * stride];
    const float a1 = acc[(long long)i * stride + 1];
    const float m  = fmaxf(a0, a1);
    const float ls = m + logf(expf(a0 - m) + expf(a1 - m));
    out[2 * i]     = a0 - ls;
    out[2 * i + 1] = a1 - ls;
}

// ---------------------------------------------------------------------------
// Launch
// ---------------------------------------------------------------------------
extern "C" void kernel_launch(void* const* d_in, const int* in_sizes, int n_in,
                              void* d_out, int out_size, void* d_ws, size_t ws_size,
                              hipStream_t stream)
{
    (void)in_sizes; (void)n_in; (void)out_size; (void)ws_size;

    const float*     x   = (const float*)d_in[0];
    const long long* ei  = (const long long*)d_in[1];     // int64 [2 x E]
    const float* Wrel[4]  = { (const float*)d_in[2], (const float*)d_in[5],
                              (const float*)d_in[8], (const float*)d_in[11] };
    const float* Wroot[4] = { (const float*)d_in[3], (const float*)d_in[6],
                              (const float*)d_in[9], (const float*)d_in[12] };
    const float* Bias[4]  = { (const float*)d_in[4], (const float*)d_in[7],
                              (const float*)d_in[10], (const float*)d_in[13] };
    float* out = (float*)d_out;

    const long long* srcIdx = ei;
    const long long* dstIdx = ei + N_EDGES;

    // layer dims: Kin, KinPad(%64), Nout, NoutPad(%16)
    const int Kin[4]   = { 1058,  500, 250, 100 };
    const int KinP[4]  = { 1088,  512, 256, 128 };
    const int Nout[4]  = {  500,  250, 100,   2 };
    const int NoutP[4] = {  512,  256, 112,  16 };

    // workspace carve-up (all 256B aligned); total ~317 MB
    size_t off = 0;
    auto carve = [&](size_t bytes) -> char* {
        char* p = (char*)d_ws + off;
        off = (off + bytes + 255) & ~(size_t)255;
        return p;
    };
    _Float16* Xh   = (_Float16*)carve((size_t)N_NODES * 1088 * sizeof(_Float16));
    float*    Hrel = (float*)   carve((size_t)N_NODES * 512  * sizeof(float));
    float*    Acc  = (float*)   carve((size_t)N_NODES * 512  * sizeof(float));
    _Float16* WrelH[4];  _Float16* WrootH[4];  float* BiasP[4];
    for (int l = 0; l < 4; ++l) {
        WrelH[l]  = (_Float16*)carve((size_t)NoutP[l] * KinP[l] * sizeof(_Float16));
        WrootH[l] = (_Float16*)carve((size_t)NoutP[l] * KinP[l] * sizeof(_Float16));
        BiasP[l]  = (float*)   carve((size_t)NoutP[l] * sizeof(float));
    }

    // 1) pad+cast weights and biases to f16 / padded f32
    for (int l = 0; l < 4; ++l) {
        dim3 wgrid((KinP[l] + 255) / 256, NoutP[l]);
        cast_pad_kernel<<<wgrid, 256, 0, stream>>>(Wrel[l],  Kin[l], Kin[l], Nout[l],
                                                   WrelH[l],  KinP[l], 0);
        cast_pad_kernel<<<wgrid, 256, 0, stream>>>(Wroot[l], Kin[l], Kin[l], Nout[l],
                                                   WrootH[l], KinP[l], 0);
        bias_pad_kernel<<<(NoutP[l] + 63) / 64, 64, 0, stream>>>(Bias[l], BiasP[l],
                                                                 Nout[l], NoutP[l]);
    }

    // 2) pad+cast input features x -> Xh [N_NODES x 1088] f16
    {
        dim3 xgrid((KinP[0] + 255) / 256, N_NODES);
        cast_pad_kernel<<<xgrid, 256, 0, stream>>>(x, Kin[0], Kin[0], N_NODES,
                                                   Xh, KinP[0], 0);
    }

    // 3) layers
    const int mTiles = N_NODES / 16;                    // 3125 exactly
    for (int l = 0; l < 4; ++l) {
        const int nTiles = NoutP[l] / 16;
        const int kIters = KinP[l] / 64;
        dim3 grid(mTiles, (nTiles + 7) / 8);
        wmma_dual_gemm<<<grid, 256, 0, stream>>>(Xh, WrelH[l], WrootH[l], BiasP[l],
                                                 Hrel, Acc, KinP[l], NoutP[l],
                                                 nTiles, kIters);

        if (Nout[l] > 4) {
            scatter_add_kernel<<<N_EDGES, 256, 0, stream>>>(Hrel, Acc, srcIdx, dstIdx,
                                                            Nout[l], NoutP[l]);
        } else {
            scatter_add_w2_kernel<<<(2 * N_EDGES + 255) / 256, 256, 0, stream>>>(
                Hrel, Acc, srcIdx, dstIdx, NoutP[l], N_EDGES);
        }

        if (l < 3) {
            // relu + cast to next layer's padded f16 input (overwrites Xh)
            dim3 rgrid((KinP[l + 1] + 255) / 256, N_NODES);
            cast_pad_kernel<<<rgrid, 256, 0, stream>>>(Acc, NoutP[l], Nout[l], N_NODES,
                                                       Xh, KinP[l + 1], 1);
        } else {
            logsoftmax2_kernel<<<(N_NODES + 255) / 256, 256, 0, stream>>>(
                Acc, out, NoutP[l], N_NODES);
        }
    }
}
